// SelfAttentionModule_44925357916293
// MI455X (gfx1250) — matrix operand order repaired and checked by reading, not compiled, not run on hardware
//
#include <hip/hip_runtime.h>
#include <hip/hip_bf16.h>
#include <math.h>

typedef __attribute__((ext_vector_type(16))) _Float16 v16h;
typedef __attribute__((ext_vector_type(8)))  float    v8f;

constexpr int BB = 2;      // batch
constexpr int CC = 64;     // channels
constexpr int C8 = 8;      // q/k channels
constexpr int NN = 9216;   // H*W

// ---------------------------------------------------------------------------
// Kernel 1: per-pixel 1x1-conv projections.
// q,k stored f16 [B,N,8] (contiguous per pixel -> b128 loads), v f16 [B,64,N].
// ---------------------------------------------------------------------------
__global__ __launch_bounds__(256) void qkv_proj_kernel(
    const float* __restrict__ x,
    const float* __restrict__ wq, const float* __restrict__ bq,
    const float* __restrict__ wk, const float* __restrict__ bk,
    const float* __restrict__ wv, const float* __restrict__ bv,
    _Float16* __restrict__ qh, _Float16* __restrict__ kh, _Float16* __restrict__ vh)
{
    __shared__ float swq[C8*CC], swk[C8*CC], swv[CC*CC];
    __shared__ float sbq[C8], sbk[C8], sbv[CC];
    const int tid = threadIdx.x;
    for (int i = tid; i < C8*CC; i += 256) { swq[i] = wq[i]; swk[i] = wk[i]; }
    for (int i = tid; i < CC*CC; i += 256) swv[i] = wv[i];
    if (tid < C8) { sbq[tid] = bq[tid]; sbk[tid] = bk[tid]; }
    if (tid < CC) sbv[tid] = bv[tid];
    __syncthreads();

    const int idx = blockIdx.x * 256 + tid;          // b*NN + n
    const int b = idx / NN, n = idx % NN;

    float xv[CC];
    const float* xp = x + (size_t)b * CC * NN + n;
    #pragma unroll
    for (int c = 0; c < CC; ++c) xv[c] = xp[(size_t)c * NN];

    #pragma unroll
    for (int o = 0; o < C8; ++o) {
        float sq = sbq[o], sk = sbk[o];
        #pragma unroll
        for (int c = 0; c < CC; ++c) {
            sq = fmaf(swq[o*CC + c], xv[c], sq);
            sk = fmaf(swk[o*CC + c], xv[c], sk);
        }
        qh[(size_t)idx * C8 + o] = (_Float16)sq;   // [B,N,8]
        kh[(size_t)idx * C8 + o] = (_Float16)sk;   // [B,N,8]
    }
    #pragma unroll 4
    for (int o = 0; o < CC; ++o) {
        float sv = sbv[o];
        #pragma unroll
        for (int c = 0; c < CC; ++c) sv = fmaf(swv[o*CC + c], xv[c], sv);
        vh[((size_t)b * CC + o) * NN + n] = (_Float16)sv;  // [B,64,N] f16
    }
}

// ---------------------------------------------------------------------------
// Kernel 2: per-row softmax stats from the same f16-rounded q/k the WMMA path
// sees. One wave per row n. Stores L[n] = max + ln(sum exp), so downstream
// P = exp(logit - L[n]) needs no multiply.
// ---------------------------------------------------------------------------
__global__ __launch_bounds__(256) void row_stats_kernel(
    const _Float16* __restrict__ qh, const _Float16* __restrict__ kh,
    float* __restrict__ Lr)
{
    const int lane = threadIdx.x & 31;
    const int wid  = blockIdx.x * 8 + (threadIdx.x >> 5);  // global row id
    const int b = wid / NN;

    float qv[C8];
    const _Float16* qp = qh + (size_t)wid * C8;
    #pragma unroll
    for (int o = 0; o < C8; ++o) qv[o] = (float)qp[o];

    const _Float16* kb = kh + (size_t)b * NN * C8;
    float mx = -INFINITY, s = 0.f;
    for (int m = lane; m < NN; m += 32) {
        const _Float16* kp = kb + (size_t)m * C8;
        float dot = 0.f;
        #pragma unroll
        for (int o = 0; o < C8; ++o) dot = fmaf((float)kp[o], qv[o], dot);
        float nm = fmaxf(mx, dot);
        s = s * __expf(mx - nm) + __expf(dot - nm);
        mx = nm;
    }
    #pragma unroll
    for (int off = 16; off > 0; off >>= 1) {
        float mo = __shfl_xor(mx, off, 32);
        float so = __shfl_xor(s,  off, 32);
        float nm = fmaxf(mx, mo);
        s = s * __expf(mx - nm) + so * __expf(mo - nm);
        mx = nm;
    }
    if (lane == 0) Lr[wid] = mx + __logf(s);
}

// ---------------------------------------------------------------------------
// Kernel 3: att[c,m] = sum_n v[c,n] * exp(q[n].k[m] - L[n]).
// Block = 128 threads = 4 waves, one 16-wide m tile, waves cover all 64 c.
// Per 128-n super-chunk:
//   produce: each wave owns a 32-n slice -> 2 logit WMMAs (K=8 zero-padded
//            to 32), exp, pack f16, ds_store_b128 into transposed LDS tile.
//   consume: after ONE barrier (ping-pong buffers), every wave runs 4 main
//            WMMAs; B frags from LDS (2x ds_load_b128), A frags = v (2x b128).
// ---------------------------------------------------------------------------
__global__ __launch_bounds__(128) void attn_wmma_kernel(
    const _Float16* __restrict__ qh, const _Float16* __restrict__ kh,
    const _Float16* __restrict__ vh, const float* __restrict__ Lr,
    float* __restrict__ att)
{
    constexpr int PSTR = 140;                   // LDS row stride (halves), padded
    __shared__ _Float16 sP[2][16][PSTR];        // [buf][m][n-local]

    const int MT = NN / 16;                     // 576 m-tiles per batch
    const int b  = blockIdx.x / MT;
    const int m_base = (blockIdx.x % MT) * 16;
    const int wave = threadIdx.x >> 5;
    const int lane = threadIdx.x & 31;
    const int half = lane >> 4;                 // 0: lanes 0-15, 1: lanes 16-31
    const int l15  = lane & 15;

    const uint4 z4 = {0u, 0u, 0u, 0u};

    // Loop-invariant logits B fragment: k column of this lane's m.
    // B layout (32x16 f16): lanes<16 hold K0..7,16..23 of column l15 (K=0..7 real,
    // 16..23 zero-pad), lanes>=16 hold K8..15,24..31 = all zero-pad.
    union { uint4 u[2]; v16h h; } Bk;
    Bk.u[0] = z4; Bk.u[1] = z4;
    if (half == 0)
        Bk.u[0] = *(const uint4*)(kh + ((size_t)b * NN + m_base + l15) * C8);

    // Main-GEMM A row: v channel for this lane.
    const _Float16* vrow = vh + ((size_t)b * CC + wave * 16 + l15) * NN;
    const float*    Lb   = Lr + (size_t)b * NN;
    const _Float16* qb   = qh + (size_t)b * NN * C8;

    v8f acc = {};

    for (int nb = 0, it = 0; nb < NN; nb += 128, ++it) {
        _Float16 (*P)[PSTR] = sP[it & 1];

        // ---- produce this wave's 32-n slice of the P tile ----
        #pragma unroll
        for (int s = 0; s < 2; ++s) {
            const int nloc = wave * 32 + s * 16;        // subtile base (local n)
            // A fragment: q rows (lanes<16: K{0..7} real + K{16..23}=0;
            // lanes>=16 all zero-pad).
            union { uint4 u[2]; v16h h; } Aq;
            Aq.u[0] = z4; Aq.u[1] = z4;
            if (half == 0)
                Aq.u[0] = *(const uint4*)(qb + (size_t)(nb + nloc + l15) * C8);

            v8f d = {};
            d = __builtin_amdgcn_wmma_f32_16x16x32_f16(
                    false, Aq.h, false, Bk.h, (short)0, d, false, false);

            // D row r (this lane) = local n (nloc + half*8 + r): grab the 8
            // matching L values with two b128 loads.
            union { uint4 u[2]; float f[8]; } Lv;
            const float* Lp = Lb + nb + nloc + half * 8;
            Lv.u[0] = *(const uint4*)(Lp);
            Lv.u[1] = *(const uint4*)(Lp + 4);

            _Float16 ph[8];
            #pragma unroll
            for (int r = 0; r < 8; ++r)
                ph[r] = (_Float16)__expf(d[r] - Lv.f[r]);

            // Transposed store: sP[m][n] so consume reads are contiguous.
            *(uint4*)(&P[l15][nloc + half * 8]) = *(const uint4*)ph;
        }

        __syncthreads();   // single barrier per iteration (ping-pong buffers)

        // ---- consume: 4 K-chunks of 32 n ----
        #pragma unroll
        for (int j = 0; j < 4; ++j) {
            const int nj = j * 32;
            union { uint4 u[2]; v16h h; } Af, Bf;
            // A = v tile: lane holds K (half*16)..(half*16+15) of row l15
            Af.u[0] = *(const uint4*)(vrow + nb + nj + half * 16);
            Af.u[1] = *(const uint4*)(vrow + nb + nj + half * 16 + 8);
            // B = P tile: lane holds K (half*16)..(half*16+15) of column l15
            Bf.u[0] = *(const uint4*)(&P[l15][nj + half * 16]);
            Bf.u[1] = *(const uint4*)(&P[l15][nj + half * 16 + 8]);
            acc = __builtin_amdgcn_wmma_f32_16x16x32_f16(
                      false, Af.h, false, Bf.h, (short)0, acc, false, false);
        }
    }

    // D layout: VGPR r, lane -> row = r + half*8, col = l15
    #pragma unroll
    for (int r = 0; r < 8; ++r) {
        const int c = wave * 16 + r + half * 8;
        att[((size_t)b * CC + c) * NN + m_base + l15] = acc[r];
    }
}

// ---------------------------------------------------------------------------
// Kernel 4: output projection out = wo @ att + bo (per-pixel 64x64 matvec).
// ---------------------------------------------------------------------------
__global__ __launch_bounds__(256) void out_proj_kernel(
    const float* __restrict__ att, const float* __restrict__ wo,
    const float* __restrict__ bo, float* __restrict__ out)
{
    __shared__ float swo[CC*CC];
    __shared__ float sbo[CC];
    const int tid = threadIdx.x;
    for (int i = tid; i < CC*CC; i += 256) swo[i] = wo[i];
    if (tid < CC) sbo[tid] = bo[tid];
    __syncthreads();

    const int idx = blockIdx.x * 256 + tid;   // b*NN + m
    const int b = idx / NN, m = idx % NN;

    float av[CC];
    const float* ap = att + (size_t)b * CC * NN + m;
    #pragma unroll
    for (int c = 0; c < CC; ++c) av[c] = ap[(size_t)c * NN];

    #pragma unroll 4
    for (int o = 0; o < CC; ++o) {
        float s = sbo[o];
        #pragma unroll
        for (int c = 0; c < CC; ++c) s = fmaf(swo[o*CC + c], av[c], s);
        out[((size_t)b * CC + o) * NN + m] = s;
    }
}

// ---------------------------------------------------------------------------
extern "C" void kernel_launch(void* const* d_in, const int* in_sizes, int n_in,
                              void* d_out, int out_size, void* d_ws, size_t ws_size,
                              hipStream_t stream)
{
    (void)in_sizes; (void)n_in; (void)out_size; (void)ws_size;
    const float* x  = (const float*)d_in[0];
    const float* wq = (const float*)d_in[1];
    const float* bq = (const float*)d_in[2];
    const float* wk = (const float*)d_in[3];
    const float* bk = (const float*)d_in[4];
    const float* wv = (const float*)d_in[5];
    const float* bv = (const float*)d_in[6];
    const float* wo = (const float*)d_in[7];
    const float* bo = (const float*)d_in[8];
    float* out = (float*)d_out;

    // Workspace carve-out (~7.4 MB total; L2-resident)
    char* p = (char*)d_ws;
    _Float16* qhp = (_Float16*)p; p += (size_t)BB * NN * C8 * sizeof(_Float16);
    _Float16* khp = (_Float16*)p; p += (size_t)BB * NN * C8 * sizeof(_Float16);
    _Float16* vhp = (_Float16*)p; p += (size_t)BB * CC * NN * sizeof(_Float16);
    float*    Lr  = (float*)p;    p += (size_t)BB * NN * sizeof(float);
    float*    att = (float*)p;

    qkv_proj_kernel<<<(BB * NN) / 256, 256, 0, stream>>>(x, wq, bq, wk, bk, wv, bv,
                                                         qhp, khp, vhp);
    row_stats_kernel<<<(BB * NN) / 8, 256, 0, stream>>>(qhp, khp, Lr);
    attn_wmma_kernel<<<BB * (NN / 16), 128, 0, stream>>>(qhp, khp, vhp, Lr, att);
    out_proj_kernel<<<(BB * NN) / 256, 256, 0, stream>>>(att, wo, bo, out);
}
